// TransformerBlock_80393197846724
// MI455X (gfx1250) — matrix operand order repaired and checked by reading, not compiled, not run on hardware
//
#include <hip/hip_runtime.h>
#include <hip/hip_bf16.h>

// ---------------------------------------------------------------------------
// Types for WMMA / TDM
// ---------------------------------------------------------------------------
typedef _Float16 v16h __attribute__((ext_vector_type(16)));
typedef _Float16 v8h  __attribute__((ext_vector_type(8)));
typedef float    v8f  __attribute__((ext_vector_type(8)));
typedef unsigned int u32x4 __attribute__((ext_vector_type(4)));
typedef int      i32x8 __attribute__((ext_vector_type(8)));
typedef int      i32x4 __attribute__((ext_vector_type(4)));

#define SHIFT_ 3
// H=W=56, WS=7, C=128, HEADS=4, HD=32, B=64, NW=64, windows=4096, rows=200704

// ---------------------------------------------------------------------------
// WMMA fragment helpers (wave32 layouts per CDNA5 ISA 7.12.2)
// A 16x32 f16: lane<16 -> row=lane, K = kb..kb+7 and kb+16..kb+23 (kb = 8*(lane>=16))
// ---------------------------------------------------------------------------
__device__ __forceinline__ v16h frag_a(const _Float16* base, int stride) {
  int lane = threadIdx.x & 31;
  const _Float16* p = base + (lane & 15) * stride + ((lane >> 4) << 3);
  v8h lo = *(const v8h*)(p);
  v8h hi = *(const v8h*)(p + 16);
  return __builtin_shufflevector(lo, hi, 0,1,2,3,4,5,6,7,8,9,10,11,12,13,14,15);
}

// B 32x16 f16 from transposed storage WT[n][k]: lane -> col n=lane&15,
// K = kb..kb+7 and kb+16..kb+23 (kb = 8*(lane>=16)).
__device__ __forceinline__ v16h frag_b(const _Float16* baseT, int strideT) {
  int lane = threadIdx.x & 31;
  const _Float16* p = baseT + (lane & 15) * strideT + ((lane >> 4) << 3);
  v8h lo = *(const v8h*)(p);
  v8h hi = *(const v8h*)(p + 16);
  return __builtin_shufflevector(lo, hi, 0,1,2,3,4,5,6,7,8,9,10,11,12,13,14,15);
}

__device__ __forceinline__ v8f wmma_f32(v16h a, v16h b, v8f c) {
  return __builtin_amdgcn_wmma_f32_16x16x32_f16(false, a, false, b, (short)0, c,
                                                false, false);
}

// ---------------------------------------------------------------------------
// Tensor Data Mover: 2D tile (rows x cols_halves of f16, contiguous rows)
// D# built per CDNA5 ISA (group0 128b, group1 256b).
// 6-arg toolchain form: (u32x4, i32x8, i32x4, i32x4, i32x8, i32 cpol)
// ---------------------------------------------------------------------------
__device__ __forceinline__ void tdm_load_rows(void* lds_dst, const void* gsrc,
                                              int rows, int cols_halves) {
  unsigned long long ga = (unsigned long long)(uintptr_t)gsrc;
  unsigned int lds = (unsigned int)(uintptr_t)lds_dst;
  u32x4 g0;
  g0[0] = 1u;                                       // count=1, is_restore=0
  g0[1] = lds;                                      // lds_addr
  g0[2] = (unsigned)(ga & 0xffffffffu);             // global_addr[31:0]
  g0[3] = (unsigned)((ga >> 32) & 0x1ffffffu) | (2u << 30); // addr[56:32] | type=2
  i32x8 g1;
  g1[0] = 1 << 16;                                  // data_size = 2 bytes
  g1[1] = (cols_halves & 0xffff) << 16;             // tensor_dim0 lo16
  g1[2] = (rows & 0xffff) << 16;                    // dim0 hi16=0 | tensor_dim1 lo16
  g1[3] = (cols_halves & 0xffff) << 16;             // dim1 hi16=0 | tile_dim0
  g1[4] = rows & 0xffff;                            // tile_dim1 | tile_dim2=0
  g1[5] = cols_halves;                              // tensor_dim0_stride lo32
  g1[6] = 0;
  g1[7] = 0;
  i32x4 z4 = {0, 0, 0, 0};
  i32x8 z8 = {0, 0, 0, 0, 0, 0, 0, 0};
  __builtin_amdgcn_tensor_load_to_lds(g0, g1, z4, z4, z8, 0);
}

// ---------------------------------------------------------------------------
// K0: weights -> f16 transposed (WT[n][k] = W[k][n])
// ---------------------------------------------------------------------------
__global__ __launch_bounds__(256) void k_prep(
    const float* __restrict__ wq, const float* __restrict__ wk,
    const float* __restrict__ wv, const float* __restrict__ wo,
    const float* __restrict__ w1, const float* __restrict__ w2,
    _Float16* __restrict__ wqT, _Float16* __restrict__ wkT,
    _Float16* __restrict__ wvT, _Float16* __restrict__ woT,
    _Float16* __restrict__ w1T, _Float16* __restrict__ w2T) {
  int tid = blockIdx.x * 256 + threadIdx.x;
  if (tid < 16384) {
    int n = tid >> 7, k = tid & 127;
    wqT[tid] = (_Float16)wq[k * 128 + n];
    wkT[tid] = (_Float16)wk[k * 128 + n];
    wvT[tid] = (_Float16)wv[k * 128 + n];
    woT[tid] = (_Float16)wo[k * 128 + n];
  }
  if (tid < 65536) {
    // w1T[512][128]: n = tid>>7, k = tid&127
    w1T[tid] = (_Float16)w1[(tid & 127) * 512 + (tid >> 7)];
    // w2T[128][512]: n = tid>>9, k = tid&511
    w2T[tid] = (_Float16)w2[(tid & 511) * 128 + (tid >> 9)];
  }
}

// ---------------------------------------------------------------------------
// K1: fused LN1 + shifted-window attention (one window per workgroup, 8 waves)
// ---------------------------------------------------------------------------
__global__ __launch_bounds__(256) void k_attn(
    const float* __restrict__ x,
    const _Float16* __restrict__ wqT, const _Float16* __restrict__ wkT,
    const _Float16* __restrict__ wvT,
    const float* __restrict__ bq, const float* __restrict__ bk,
    const float* __restrict__ bv,
    const float* __restrict__ rel, const float* __restrict__ lng,
    const float* __restrict__ lnb,
    _Float16* __restrict__ oG) {
  __shared__ __align__(16) float    sS[64 * 64];    // scores f32 / p f16 / sH f16
  __shared__ __align__(16) _Float16 sQ[64 * 128];
  __shared__ __align__(16) _Float16 sK[64 * 128];
  __shared__ __align__(16) _Float16 sVT[128 * 64];  // v transposed [dim][token]
  _Float16* sH = (_Float16*)sS;                     // 64x128 f16 (16KB overlay)

  const int tid = threadIdx.x;
  const int lane = tid & 31;
  const int wave = tid >> 5;
  const int w = blockIdx.x;
  const int bb = w >> 6;
  const int widx = w & 63;
  const int wy = widx >> 3;
  const int wx = widx & 7;

  // zero pad rows 49..63 of sH
  for (int i = tid; i < 15 * 128; i += 256) sH[49 * 128 + i] = (_Float16)0.f;

  // LN1 + roll(-3,-3) + window gather -> sH (f16)
  for (int r = wave; r < 49; r += 8) {
    int ty = r / 7, tx = r % 7;
    int y = (wy * 7 + ty + SHIFT_) % 56;
    int xx = (wx * 7 + tx + SHIFT_) % 56;
    const float* src = x + ((size_t)bb * 3136 + y * 56 + xx) * 128;
    float4 v = *(const float4*)(src + lane * 4);
    float s = v.x + v.y + v.z + v.w;
    float s2 = v.x * v.x + v.y * v.y + v.z * v.z + v.w * v.w;
    for (int off = 16; off >= 1; off >>= 1) {
      s += __shfl_xor(s, off, 32);
      s2 += __shfl_xor(s2, off, 32);
    }
    float mean = s * (1.f / 128.f);
    float var = s2 * (1.f / 128.f) - mean * mean;
    float rs = rsqrtf(var + 1e-5f);
    float vv[4] = {v.x, v.y, v.z, v.w};
    for (int i = 0; i < 4; ++i) {
      int c = lane * 4 + i;
      sH[r * 128 + c] = (_Float16)((vv[i] - mean) * rs * lng[c] + lnb[c]);
    }
  }
  __syncthreads();

  // QKV: 96 16x16 tiles, 12 per wave; q pre-scaled by 1/sqrt(32)
  for (int t = wave; t < 96; t += 8) {
    int mat = t / 32, rem = t % 32;
    int mi = rem >> 3, ni = rem & 7;
    const _Float16* wT = (mat == 0) ? wqT : (mat == 1) ? wkT : wvT;
    const float* bias = (mat == 0) ? bq : (mat == 1) ? bk : bv;
    v8f acc = {0.f, 0.f, 0.f, 0.f, 0.f, 0.f, 0.f, 0.f};
    for (int kk = 0; kk < 4; ++kk) {
      v16h a = frag_a(sH + mi * 16 * 128 + kk * 32, 128);
      v16h bf = frag_b(wT + (ni * 16) * 128 + kk * 32, 128);
      acc = wmma_f32(a, bf, acc);
    }
    int n = ni * 16 + (lane & 15);
    int mb = mi * 16 + ((lane >> 4) << 3);
    float bn = bias[n];
    if (mat == 0) {
      for (int j = 0; j < 8; ++j)
        sQ[(mb + j) * 128 + n] = (_Float16)((acc[j] + bn) * 0.17677669529663687f);
    } else if (mat == 1) {
      for (int j = 0; j < 8; ++j) sK[(mb + j) * 128 + n] = (_Float16)(acc[j] + bn);
    } else {
      for (int j = 0; j < 8; ++j) sVT[n * 64 + (mb + j)] = (_Float16)(acc[j] + bn);
    }
  }
  __syncthreads();

  for (int h = 0; h < 4; ++h) {
    // scores + rel-pos bias + shift mask
    for (int t = wave; t < 16; t += 8) {
      int mi = t >> 2, ni = t & 3;
      v16h a = frag_a(sQ + mi * 16 * 128 + h * 32, 128);
      v16h bf = frag_b(sK + ni * 16 * 128 + h * 32, 128);
      v8f acc = {0.f, 0.f, 0.f, 0.f, 0.f, 0.f, 0.f, 0.f};
      acc = wmma_f32(a, bf, acc);
      int n = ni * 16 + (lane & 15);
      int mb = mi * 16 + ((lane >> 4) << 3);
      bool kvalid = n < 49;
      int ky = kvalid ? n / 7 : 0, kx = kvalid ? n % 7 : 0;
      int Yk = wy * 7 + ky, Xk = wx * 7 + kx;
      int lk = (Yk < 49 ? 0 : (Yk < 53 ? 1 : 2)) * 3 + (Xk < 49 ? 0 : (Xk < 53 ? 1 : 2));
      for (int j = 0; j < 8; ++j) {
        int m = mb + j;
        float sv;
        if (!kvalid) {
          sv = -1e30f;
        } else if (m < 49) {
          int qy = m / 7, qx = m % 7;
          int dy = (ky - qy + 13) % 13;
          int dx = (kx - qx + 13) % 13;
          sv = acc[j] + rel[(dy * 13 + dx) * 4 + h];
          int Yq = wy * 7 + qy, Xq = wx * 7 + qx;
          int lq = (Yq < 49 ? 0 : (Yq < 53 ? 1 : 2)) * 3 +
                   (Xq < 49 ? 0 : (Xq < 53 ? 1 : 2));
          if (lq != lk) sv -= 100.f;
        } else {
          sv = acc[j];  // padded q row, discarded later
        }
        sS[m * 64 + n] = sv;
      }
    }
    __syncthreads();
    // softmax, p written f16 in place over the f32 row (write ptr trails read ptr)
    if (tid < 64) {
      float* row = sS + tid * 64;
      _Float16* prow = (_Float16*)row;
      float mx = -1e30f;
      for (int c = 0; c < 64; ++c) mx = fmaxf(mx, row[c]);
      float sum = 0.f;
      for (int c = 0; c < 64; ++c) sum += __expf(row[c] - mx);
      float inv = 1.f / sum;
      for (int c = 0; c < 64; ++c) {
        float e = __expf(row[c] - mx) * inv;
        prow[c] = (_Float16)e;
      }
    }
    __syncthreads();
    // o_h = p @ v_h  (8 tiles, one per wave); p rows stride 128 halves (256B rows)
    {
      int mi = wave >> 1, ni = wave & 1;
      const _Float16* pbuf = (const _Float16*)sS;
      v8f acc = {0.f, 0.f, 0.f, 0.f, 0.f, 0.f, 0.f, 0.f};
      for (int kk = 0; kk < 2; ++kk) {
        v16h a = frag_a(pbuf + mi * 16 * 128 + kk * 32, 128);
        v16h bf = frag_b(sVT + (h * 32 + ni * 16) * 64 + kk * 32, 64);
        acc = wmma_f32(a, bf, acc);
      }
      int n = ni * 16 + (lane & 15);
      int mb = mi * 16 + ((lane >> 4) << 3);
      for (int j = 0; j < 8; ++j)
        oG[(size_t)w * 8192 + (size_t)(mb + j) * 128 + h * 32 + n] = (_Float16)acc[j];
    }
    __syncthreads();
  }
}

// ---------------------------------------------------------------------------
// K2: output projection + shortcut, un-window/un-roll scatter. TDM stages tile.
// ---------------------------------------------------------------------------
__global__ __launch_bounds__(256) void k_proj(
    const _Float16* __restrict__ oG, const _Float16* __restrict__ woT,
    const float* __restrict__ bo, const float* __restrict__ x,
    float* __restrict__ hres) {
  __shared__ __align__(16) _Float16 sA[64 * 128];
  const int tid = threadIdx.x, lane = tid & 31, wave = tid >> 5;
  const int w = blockIdx.x, bb = w >> 6, widx = w & 63;
  const int wy = widx >> 3, wx = widx & 7;
  if (tid < 32) {
    tdm_load_rows(sA, oG + (size_t)w * 8192, 64, 128);
    __builtin_amdgcn_s_wait_tensorcnt(0);
  }
  __syncthreads();
  for (int t = wave; t < 32; t += 8) {
    int mi = t >> 3, ni = t & 7;
    v8f acc = {0.f, 0.f, 0.f, 0.f, 0.f, 0.f, 0.f, 0.f};
    for (int kk = 0; kk < 4; ++kk) {
      v16h a = frag_a(sA + mi * 16 * 128 + kk * 32, 128);
      v16h bf = frag_b(woT + (ni * 16) * 128 + kk * 32, 128);
      acc = wmma_f32(a, bf, acc);
    }
    int n = ni * 16 + (lane & 15);
    int mb = mi * 16 + ((lane >> 4) << 3);
    float bn = bo[n];
    for (int j = 0; j < 8; ++j) {
      int m = mb + j;
      if (m < 49) {
        int ty = m / 7, tx = m % 7;
        int y = (wy * 7 + ty + SHIFT_) % 56;
        int xx = (wx * 7 + tx + SHIFT_) % 56;
        size_t idx = ((size_t)bb * 3136 + y * 56 + xx) * 128 + n;
        hres[idx] = x[idx] + acc[j] + bn;
      }
    }
  }
}

// ---------------------------------------------------------------------------
// K3: LayerNorm (row per wave) -> f16
// ---------------------------------------------------------------------------
__global__ __launch_bounds__(256) void k_ln(const float* __restrict__ in,
                                            const float* __restrict__ g,
                                            const float* __restrict__ b,
                                            _Float16* __restrict__ outh) {
  int row = blockIdx.x * 8 + (threadIdx.x >> 5);
  int lane = threadIdx.x & 31;
  const float4 v = *(const float4*)(in + (size_t)row * 128 + lane * 4);
  float s = v.x + v.y + v.z + v.w;
  float s2 = v.x * v.x + v.y * v.y + v.z * v.z + v.w * v.w;
  for (int off = 16; off >= 1; off >>= 1) {
    s += __shfl_xor(s, off, 32);
    s2 += __shfl_xor(s2, off, 32);
  }
  float mean = s * (1.f / 128.f);
  float var = s2 * (1.f / 128.f) - mean * mean;
  float rs = rsqrtf(var + 1e-5f);
  float vv[4] = {v.x, v.y, v.z, v.w};
  for (int i = 0; i < 4; ++i) {
    int c = lane * 4 + i;
    outh[(size_t)row * 128 + c] = (_Float16)((vv[i] - mean) * rs * g[c] + b[c]);
  }
}

// ---------------------------------------------------------------------------
// K4: fused MLP (128->512 exact GELU ->128) + residual -> d_out. TDM-staged A.
// ---------------------------------------------------------------------------
__global__ __launch_bounds__(256) void k_mlp(
    const _Float16* __restrict__ m16, const _Float16* __restrict__ w1T,
    const float* __restrict__ b1, const _Float16* __restrict__ w2T,
    const float* __restrict__ b2, const float* __restrict__ hres,
    float* __restrict__ out) {
  __shared__ __align__(16) _Float16 sA[32 * 128];     // 8KB
  __shared__ __align__(16) _Float16 sHid[32 * 512];   // 32KB
  const int tid = threadIdx.x, lane = tid & 31, wave = tid >> 5;
  const int r0 = blockIdx.x * 32;
  if (tid < 32) {
    tdm_load_rows(sA, m16 + (size_t)r0 * 128, 32, 128);
    __builtin_amdgcn_s_wait_tensorcnt(0);
  }
  __syncthreads();
  // stage 1: 2x32 tiles, K=128
  for (int t = wave; t < 64; t += 8) {
    int mi = t >> 5, ni = t & 31;
    v8f acc = {0.f, 0.f, 0.f, 0.f, 0.f, 0.f, 0.f, 0.f};
    for (int kk = 0; kk < 4; ++kk) {
      v16h a = frag_a(sA + mi * 16 * 128 + kk * 32, 128);
      v16h bf = frag_b(w1T + (ni * 16) * 128 + kk * 32, 128);
      acc = wmma_f32(a, bf, acc);
    }
    int n = ni * 16 + (lane & 15);
    int mb = mi * 16 + ((lane >> 4) << 3);
    float bn = b1[n];
    for (int j = 0; j < 8; ++j) {
      float vv = acc[j] + bn;
      float ge = 0.5f * vv * (1.f + erff(vv * 0.70710678118654752f));
      sHid[(mb + j) * 512 + n] = (_Float16)ge;
    }
  }
  __syncthreads();
  // stage 2: 2x8 tiles, K=512
  for (int t = wave; t < 16; t += 8) {
    int mi = t >> 3, ni = t & 7;
    v8f acc = {0.f, 0.f, 0.f, 0.f, 0.f, 0.f, 0.f, 0.f};
    for (int kk = 0; kk < 16; ++kk) {
      v16h a = frag_a(sHid + mi * 16 * 512 + kk * 32, 512);
      v16h bf = frag_b(w2T + (ni * 16) * 512 + kk * 32, 512);
      acc = wmma_f32(a, bf, acc);
    }
    int n = ni * 16 + (lane & 15);
    int mb = mi * 16 + ((lane >> 4) << 3);
    float bn = b2[n];
    for (int j = 0; j < 8; ++j) {
      size_t row = (size_t)(r0 + mb + j);
      out[row * 128 + n] = hres[row * 128 + n] + acc[j] + bn;
    }
  }
}

// ---------------------------------------------------------------------------
extern "C" void kernel_launch(void* const* d_in, const int* in_sizes, int n_in,
                              void* d_out, int out_size, void* d_ws, size_t ws_size,
                              hipStream_t stream) {
  (void)in_sizes; (void)n_in; (void)out_size; (void)ws_size;
  const float* x    = (const float*)d_in[0];
  const float* wq   = (const float*)d_in[1];
  const float* bq   = (const float*)d_in[2];
  const float* wk   = (const float*)d_in[3];
  const float* bk   = (const float*)d_in[4];
  const float* wv   = (const float*)d_in[5];
  const float* bv   = (const float*)d_in[6];
  const float* wo   = (const float*)d_in[7];
  const float* bo   = (const float*)d_in[8];
  const float* rel  = (const float*)d_in[9];
  const float* ln1g = (const float*)d_in[10];
  const float* ln1b = (const float*)d_in[11];
  const float* ln2g = (const float*)d_in[12];
  const float* ln2b = (const float*)d_in[13];
  const float* w1   = (const float*)d_in[14];
  const float* b1   = (const float*)d_in[15];
  const float* w2   = (const float*)d_in[16];
  const float* b2   = (const float*)d_in[17];

  char* ws = (char*)d_ws;
  size_t o = 0;
  _Float16* wqT = (_Float16*)(ws + o); o += 32768;
  _Float16* wkT = (_Float16*)(ws + o); o += 32768;
  _Float16* wvT = (_Float16*)(ws + o); o += 32768;
  _Float16* woT = (_Float16*)(ws + o); o += 32768;
  _Float16* w1T = (_Float16*)(ws + o); o += 131072;
  _Float16* w2T = (_Float16*)(ws + o); o += 131072;
  _Float16* oG  = (_Float16*)(ws + o); o += (size_t)4096 * 64 * 128 * 2;   // 64 MB
  float*    hres = (float*)(ws + o);   o += (size_t)200704 * 128 * 4;      // 98 MB
  _Float16* m16 = (_Float16*)(ws + o); o += (size_t)200704 * 128 * 2;      // 49 MB

  k_prep<<<256, 256, 0, stream>>>(wq, wk, wv, wo, w1, w2,
                                  wqT, wkT, wvT, woT, w1T, w2T);
  k_attn<<<4096, 256, 0, stream>>>(x, wqT, wkT, wvT, bq, bk, bv,
                                   rel, ln1g, ln1b, oG);
  k_proj<<<4096, 256, 0, stream>>>(oG, woT, bo, x, hres);
  k_ln<<<25088, 256, 0, stream>>>(hres, ln2g, ln2b, m16);
  k_mlp<<<6272, 256, 0, stream>>>(m16, w1T, b1, w2T, b2, hres, (float*)d_out);
}
